// SelfAttention_68607807586666
// MI455X (gfx1250) — matrix operand order repaired and checked by reading, not compiled, not run on hardware
//
#include <hip/hip_runtime.h>
#include <hip/hip_bf16.h>

// ---- problem constants ----
#define BB   4
#define TT   2048
#define EE   768
#define HH   12
#define HD   64
#define E3   2304

typedef __attribute__((ext_vector_type(16))) _Float16 v16h;
typedef __attribute__((ext_vector_type(8)))  _Float16 v8h;
typedef __attribute__((ext_vector_type(8)))  float    v8f;

static __device__ __forceinline__ v16h make_v16h(v8h a, v8h b) {
    v16h r;
#pragma unroll
    for (int i = 0; i < 8; ++i) { r[i] = a[i]; r[8 + i] = b[i]; }
    return r;
}

// A 16x32 f16 fragment (ISA 7.12.2): lane<16: M=lane, K={0..7,16..23};
// lane>=16: M=lane-16, K={8..15,24..31}. rowPtr -> 32 contiguous K-halves.
static __device__ __forceinline__ v16h frag_a(const _Float16* rowPtr, int lane) {
    const int koff = (lane & 16) ? 8 : 0;
    v8h lo = *(const v8h*)(rowPtr + koff);
    v8h hi = *(const v8h*)(rowPtr + 16 + koff);
    return make_v16h(lo, hi);
}

// B 32x16 f16 fragment: lane<16: K={0..7,16..23}; lane>=16: K={8..15,24..31}, N = lane%16.
// colPtr -> 32 contiguous K-halves of this lane's N column.
static __device__ __forceinline__ v16h frag_b(const _Float16* colPtr, int lane) {
    const int koff = (lane & 16) ? 8 : 0;
    v8h lo = *(const v8h*)(colPtr + koff);
    v8h hi = *(const v8h*)(colPtr + 16 + koff);
    return make_v16h(lo, hi);
}

static __device__ __forceinline__ v8f wmma16(v16h a, v16h b, v8f c) {
    return __builtin_amdgcn_wmma_f32_16x16x32_f16(false, a, false, b,
                                                  (short)0, c, false, false);
}

// Async DMA: global -> LDS, 16B per lane, tracked by ASYNCcnt (CDNA5).
static __device__ __forceinline__ void async_copy_b128(const void* gsrc,
                                                       const _Float16* ldsDst) {
    const unsigned long long ga = (unsigned long long)(uintptr_t)gsrc;
    const unsigned lo = (unsigned)(uintptr_t)ldsDst;   // LDS aperture: addr[31:0]
    asm volatile("global_load_async_to_lds_b128 %0, %1, off"
                 :: "v"(lo), "v"(ga) : "memory");
}
static __device__ __forceinline__ void wait_asynccnt0() {
    asm volatile("s_wait_asynccnt 0x0" ::: "memory");
}

// =====================================================================
// Kernel 1: qkv = x @ Wqkv -> Q/K/V f16 [B,H,T,hd], Q pre-scaled 0.125.
// Macro tile 128x64, 8 waves (4x2) each 32x32 (4 wmma/K-step),
// BK=32, LDS double-buffered, 1 barrier/iter.
// =====================================================================
#define AS_STRIDE 40
#define AS_BUF (128 * AS_STRIDE)
#define BS_BUF (64 * AS_STRIDE)

__global__ __launch_bounds__(256) void qkv_gemm_kernel(
    const float* __restrict__ x, const float* __restrict__ w,
    _Float16* __restrict__ Qh, _Float16* __restrict__ Kh, _Float16* __restrict__ Vh)
{
    __shared__ alignas(16) _Float16 As[2 * AS_BUF];   // [buf][m][k] stride 40
    __shared__ alignas(16) _Float16 Bs[2 * BS_BUF];   // [buf][n][k] (transposed)

    const int tid   = threadIdx.x;
    const int lane  = tid & 31;
    const int wv    = tid >> 5;
    const int waveM = wv & 3;             // 4 row-waves * 32 rows
    const int waveN = wv >> 2;            // 2 col-waves * 32 cols
    const int rowBase = blockIdx.y * 128;
    const int colBase = blockIdx.x * 64;
    const int m = lane & 15;

    auto stage = [&](int k0, int buf) {
        {   // A tile 128x32 f32->f16 (16 floats/thread)
            const int r = tid >> 1;
            const int c = (tid & 1) * 16;
            const float* gp = x + (size_t)(rowBase + r) * EE + k0 + c;
            if (k0 + 64 < EE) __builtin_prefetch(gp + 64, 0, 0);  // global_prefetch_b8
            const float4* p = (const float4*)gp;
            float4 f0 = p[0], f1 = p[1], f2 = p[2], f3 = p[3];
            v8h h0, h1;
            h0[0]=(_Float16)f0.x; h0[1]=(_Float16)f0.y; h0[2]=(_Float16)f0.z; h0[3]=(_Float16)f0.w;
            h0[4]=(_Float16)f1.x; h0[5]=(_Float16)f1.y; h0[6]=(_Float16)f1.z; h0[7]=(_Float16)f1.w;
            h1[0]=(_Float16)f2.x; h1[1]=(_Float16)f2.y; h1[2]=(_Float16)f2.z; h1[3]=(_Float16)f2.w;
            h1[4]=(_Float16)f3.x; h1[5]=(_Float16)f3.y; h1[6]=(_Float16)f3.z; h1[7]=(_Float16)f3.w;
            _Float16* dst = As + buf * AS_BUF + r * AS_STRIDE + c;
            *(v8h*)(dst) = h0;
            *(v8h*)(dst + 8) = h1;
        }
        {   // B tile 32x64 f32->f16, transposed store [n][k]
            const int kr = tid >> 3;
            const int nc = (tid & 7) * 8;
            const float* gp = w + (size_t)(k0 + kr) * E3 + colBase + nc;
            if (k0 + 64 < EE) __builtin_prefetch(gp + (size_t)32 * E3, 0, 0);
            const float4* p = (const float4*)gp;
            float4 f0 = p[0], f1 = p[1];
            _Float16* bb = Bs + buf * BS_BUF;
            bb[(nc + 0) * AS_STRIDE + kr] = (_Float16)f0.x;
            bb[(nc + 1) * AS_STRIDE + kr] = (_Float16)f0.y;
            bb[(nc + 2) * AS_STRIDE + kr] = (_Float16)f0.z;
            bb[(nc + 3) * AS_STRIDE + kr] = (_Float16)f0.w;
            bb[(nc + 4) * AS_STRIDE + kr] = (_Float16)f1.x;
            bb[(nc + 5) * AS_STRIDE + kr] = (_Float16)f1.y;
            bb[(nc + 6) * AS_STRIDE + kr] = (_Float16)f1.z;
            bb[(nc + 7) * AS_STRIDE + kr] = (_Float16)f1.w;
        }
    };

    v8f acc00 = {}, acc01 = {}, acc10 = {}, acc11 = {};

    stage(0, 0);
    __syncthreads();
    int cur = 0;
    for (int k0 = 0; k0 < EE; k0 += 32) {
        if (k0 + 32 < EE) stage(k0 + 32, cur ^ 1);
        const _Float16* as = As + cur * AS_BUF;
        const _Float16* bs = Bs + cur * BS_BUF;
        v16h a0 = frag_a(as + (waveM * 32 +      m) * AS_STRIDE, lane);
        v16h a1 = frag_a(as + (waveM * 32 + 16 + m) * AS_STRIDE, lane);
        v16h b0 = frag_b(bs + (waveN * 32 +      m) * AS_STRIDE, lane);
        v16h b1 = frag_b(bs + (waveN * 32 + 16 + m) * AS_STRIDE, lane);
        acc00 = wmma16(a0, b0, acc00);
        acc01 = wmma16(a0, b1, acc01);
        acc10 = wmma16(a1, b0, acc10);
        acc11 = wmma16(a1, b1, acc11);
        __syncthreads();
        cur ^= 1;
    }

    // epilogue: C layout VGPR r: lane<16 -> M=r, lane>=16 -> M=r+8
    const int rOfs = (lane & 16) ? 8 : 0;
#pragma unroll
    for (int cb = 0; cb < 2; ++cb) {
        const int col0 = colBase + waveN * 32 + cb * 16;  // within one head
        const int sel  = col0 / EE;
        const int e0   = col0 % EE;
        const int hh   = e0 >> 6;
        const int d0   = e0 & 63;
        _Float16* dst  = (sel == 0) ? Qh : ((sel == 1) ? Kh : Vh);
        const float scale = (sel == 0) ? 0.125f : 1.0f;
#pragma unroll
        for (int rt = 0; rt < 2; ++rt) {
            v8f acc = (rt == 0) ? (cb == 0 ? acc00 : acc01)
                                : (cb == 0 ? acc10 : acc11);
#pragma unroll
            for (int r = 0; r < 8; ++r) {
                const int gr = rowBase + waveM * 32 + rt * 16 + r + rOfs;
                const int bidx = gr >> 11;
                const int t    = gr & (TT - 1);
                dst[(((size_t)(bidx * HH + hh) * TT + t) * HD) + d0 + m] =
                    (_Float16)(acc[r] * scale);
            }
        }
    }
}

// =====================================================================
// Kernel 2: flash attention. Block = (b,h,128 q-rows); 8 waves x 16 rows.
// 32-key chunks, double-buffered. K staged via async DMA (ASYNCcnt),
// V staged transposed [d][key] manually.
// =====================================================================
#define KS_BUF (32 * 64)
#define VT_BUF (64 * 40)

__global__ __launch_bounds__(256) void attn_kernel(
    const _Float16* __restrict__ Qh, const _Float16* __restrict__ Kh,
    const _Float16* __restrict__ Vh, _Float16* __restrict__ Oh)
{
    __shared__ alignas(16) _Float16 Ks[2 * KS_BUF];
    __shared__ alignas(16) _Float16 VT[2 * VT_BUF];
    __shared__ alignas(16) _Float16 Pt[8 * 16 * 40];

    const int tid  = threadIdx.x;
    const int lane = tid & 31;
    const int wv   = tid >> 5;
    const int m    = lane & 15;
    const int rOfs = (lane & 16) ? 8 : 0;

    const int q0 = blockIdx.x * 128;
    const int hh = blockIdx.y;
    const int b  = blockIdx.z;
    const size_t bh = ((size_t)b * HH + hh) * TT;
    const int qBase = q0 + wv * 16;

    auto stage = [&](int kc, int buf) {
        const int k0  = kc * 32;
        const int key = tid >> 3;
        const int d   = (tid & 7) * 8;
        // K: one async b128 per lane, natural [key][d] layout (4KB total)
        async_copy_b128(Kh + (bh + k0 + key) * HD + d,
                        Ks + buf * KS_BUF + key * 64 + d);
        // V: manual transpose into [d][key]
        v8h vv = *(const v8h*)(Vh + (bh + k0 + key) * HD + d);
#pragma unroll
        for (int j = 0; j < 8; ++j)
            VT[buf * VT_BUF + (d + j) * 40 + key] = vv[j];
    };

    const _Float16* qrow = Qh + (bh + qBase + m) * HD;
    v16h qa0 = frag_a(qrow, lane);
    v16h qa1 = frag_a(qrow + 32, lane);

    v8f o0 = {}, o1 = {}, o2 = {}, o3 = {};
    float mrow[8], lrow[8];
#pragma unroll
    for (int r = 0; r < 8; ++r) { mrow[r] = -1.0e30f; lrow[r] = 0.0f; }

    const int nChunks = (q0 + 128) / 32;
    stage(0, 0);
    wait_asynccnt0();          // own async copies done ...
    __syncthreads();           // ... then publish to the workgroup
    int cur = 0;
    for (int kc = 0; kc < nChunks; ++kc) {
        const int k0 = kc * 32;
        if (kc + 1 < nChunks) stage(kc + 1, cur ^ 1);

        if (k0 <= qBase + 15) {              // wave-uniform: EXEC stays full
            const _Float16* ks = Ks + cur * KS_BUF;
            const _Float16* vt = VT + cur * VT_BUF;
            v8f s0 = {}, s1 = {};
            s0 = wmma16(qa0, frag_b(ks + m * 64,             lane), s0);
            s0 = wmma16(qa1, frag_b(ks + m * 64 + 32,        lane), s0);
            s1 = wmma16(qa0, frag_b(ks + (16 + m) * 64,      lane), s1);
            s1 = wmma16(qa1, frag_b(ks + (16 + m) * 64 + 32, lane), s1);

            if (k0 + 31 > qBase) {           // causal mask near diagonal
#pragma unroll
                for (int r = 0; r < 8; ++r) {
                    const int qr = qBase + r + rOfs;
                    if (k0 + m > qr)      s0[r] = -1.0e30f;
                    if (k0 + 16 + m > qr) s1[r] = -1.0e30f;
                }
            }

            _Float16* pt = Pt + wv * (16 * 40);
#pragma unroll
            for (int r = 0; r < 8; ++r) {
                float mx = fmaxf(s0[r], s1[r]);
                mx = fmaxf(mx, __shfl_xor(mx, 1, 32));
                mx = fmaxf(mx, __shfl_xor(mx, 2, 32));
                mx = fmaxf(mx, __shfl_xor(mx, 4, 32));
                mx = fmaxf(mx, __shfl_xor(mx, 8, 32));
                const float mnew  = fmaxf(mrow[r], mx);
                const float alpha = __expf(mrow[r] - mnew);
                const float p0 = __expf(s0[r] - mnew);
                const float p1 = __expf(s1[r] - mnew);
                float ps = p0 + p1;
                ps += __shfl_xor(ps, 1, 32);
                ps += __shfl_xor(ps, 2, 32);
                ps += __shfl_xor(ps, 4, 32);
                ps += __shfl_xor(ps, 8, 32);
                lrow[r] = lrow[r] * alpha + ps;
                mrow[r] = mnew;
                o0[r] *= alpha; o1[r] *= alpha; o2[r] *= alpha; o3[r] *= alpha;
                pt[(r + rOfs) * 40 + m]      = (_Float16)p0;
                pt[(r + rOfs) * 40 + 16 + m] = (_Float16)p1;
            }

            v16h pa = frag_a(pt + m * 40, lane);   // per-wave LDS, in-order
            o0 = wmma16(pa, frag_b(vt + (0  + m) * 40, lane), o0);
            o1 = wmma16(pa, frag_b(vt + (16 + m) * 40, lane), o1);
            o2 = wmma16(pa, frag_b(vt + (32 + m) * 40, lane), o2);
            o3 = wmma16(pa, frag_b(vt + (48 + m) * 40, lane), o3);
        }
        wait_asynccnt0();      // K chunk for next iter landed in LDS
        __syncthreads();
        cur ^= 1;
    }

#pragma unroll
    for (int r = 0; r < 8; ++r) {
        const float inv = 1.0f / lrow[r];
        const int t = qBase + r + rOfs;
        const size_t base = ((size_t)b * TT + t) * EE + hh * HD;
        Oh[base +  0 + m] = (_Float16)(o0[r] * inv);
        Oh[base + 16 + m] = (_Float16)(o1[r] * inv);
        Oh[base + 32 + m] = (_Float16)(o2[r] * inv);
        Oh[base + 48 + m] = (_Float16)(o3[r] * inv);
    }
}

// =====================================================================
// Kernel 3: out = attnOut(f16) @ Wproj -> f32 [B,T,E]. Same 128x64 tiling.
// =====================================================================
__global__ __launch_bounds__(256) void proj_gemm_kernel(
    const _Float16* __restrict__ A, const float* __restrict__ w,
    float* __restrict__ out)
{
    __shared__ alignas(16) _Float16 As[2 * AS_BUF];
    __shared__ alignas(16) _Float16 Bs[2 * BS_BUF];

    const int tid   = threadIdx.x;
    const int lane  = tid & 31;
    const int wv    = tid >> 5;
    const int waveM = wv & 3;
    const int waveN = wv >> 2;
    const int rowBase = blockIdx.y * 128;
    const int colBase = blockIdx.x * 64;
    const int m = lane & 15;

    auto stage = [&](int k0, int buf) {
        {   // A tile 128x32 f16 (16 halves/thread)
            const int r = tid >> 1;
            const int c = (tid & 1) * 16;
            const _Float16* gp = A + (size_t)(rowBase + r) * EE + k0 + c;
            if (k0 + 64 < EE) __builtin_prefetch(gp + 64, 0, 0);
            const v8h* p = (const v8h*)gp;
            v8h h0 = p[0], h1 = p[1];
            _Float16* dst = As + buf * AS_BUF + r * AS_STRIDE + c;
            *(v8h*)(dst) = h0;
            *(v8h*)(dst + 8) = h1;
        }
        {   // B tile 32x64 f32->f16 transposed
            const int kr = tid >> 3;
            const int nc = (tid & 7) * 8;
            const float* gp = w + (size_t)(k0 + kr) * EE + colBase + nc;
            if (k0 + 64 < EE) __builtin_prefetch(gp + (size_t)32 * EE, 0, 0);
            const float4* p = (const float4*)gp;
            float4 f0 = p[0], f1 = p[1];
            _Float16* bb = Bs + buf * BS_BUF;
            bb[(nc + 0) * AS_STRIDE + kr] = (_Float16)f0.x;
            bb[(nc + 1) * AS_STRIDE + kr] = (_Float16)f0.y;
            bb[(nc + 2) * AS_STRIDE + kr] = (_Float16)f0.z;
            bb[(nc + 3) * AS_STRIDE + kr] = (_Float16)f0.w;
            bb[(nc + 4) * AS_STRIDE + kr] = (_Float16)f1.x;
            bb[(nc + 5) * AS_STRIDE + kr] = (_Float16)f1.y;
            bb[(nc + 6) * AS_STRIDE + kr] = (_Float16)f1.z;
            bb[(nc + 7) * AS_STRIDE + kr] = (_Float16)f1.w;
        }
    };

    v8f acc00 = {}, acc01 = {}, acc10 = {}, acc11 = {};

    stage(0, 0);
    __syncthreads();
    int cur = 0;
    for (int k0 = 0; k0 < EE; k0 += 32) {
        if (k0 + 32 < EE) stage(k0 + 32, cur ^ 1);
        const _Float16* as = As + cur * AS_BUF;
        const _Float16* bs = Bs + cur * BS_BUF;
        v16h a0 = frag_a(as + (waveM * 32 +      m) * AS_STRIDE, lane);
        v16h a1 = frag_a(as + (waveM * 32 + 16 + m) * AS_STRIDE, lane);
        v16h b0 = frag_b(bs + (waveN * 32 +      m) * AS_STRIDE, lane);
        v16h b1 = frag_b(bs + (waveN * 32 + 16 + m) * AS_STRIDE, lane);
        acc00 = wmma16(a0, b0, acc00);
        acc01 = wmma16(a0, b1, acc01);
        acc10 = wmma16(a1, b0, acc10);
        acc11 = wmma16(a1, b1, acc11);
        __syncthreads();
        cur ^= 1;
    }

    const int rOfs = (lane & 16) ? 8 : 0;
#pragma unroll
    for (int cb = 0; cb < 2; ++cb) {
        const int col = colBase + waveN * 32 + cb * 16 + m;
#pragma unroll
        for (int rt = 0; rt < 2; ++rt) {
            v8f acc = (rt == 0) ? (cb == 0 ? acc00 : acc01)
                                : (cb == 0 ? acc10 : acc11);
#pragma unroll
            for (int r = 0; r < 8; ++r) {
                const int gr = rowBase + waveM * 32 + rt * 16 + r + rOfs;
                out[(size_t)gr * EE + col] = acc[r];
            }
        }
    }
}

// =====================================================================
extern "C" void kernel_launch(void* const* d_in, const int* in_sizes, int n_in,
                              void* d_out, int out_size, void* d_ws, size_t ws_size,
                              hipStream_t stream) {
    const float* x     = (const float*)d_in[0];   // [4,2048,768]
    const float* wqkv  = (const float*)d_in[1];   // [768,2304]
    const float* wproj = (const float*)d_in[2];   // [768,768]
    float* out = (float*)d_out;                   // [4,2048,768]

    const size_t elems = (size_t)BB * HH * TT * HD;   // 6,291,456
    _Float16* Qh = (_Float16*)d_ws;
    _Float16* Kh = Qh + elems;
    _Float16* Vh = Kh + elems;
    _Float16* Ao = Vh + elems;    // attn output f16 [B,T,E]

    dim3 g1(E3 / 64, (BB * TT) / 128);   // (36, 64)
    qkv_gemm_kernel<<<g1, 256, 0, stream>>>(x, wqkv, Qh, Kh, Vh);

    dim3 g2(TT / 128, HH, BB);           // (16, 12, 4)
    attn_kernel<<<g2, 256, 0, stream>>>(Qh, Kh, Vh, Ao);

    dim3 g3(EE / 64, (BB * TT) / 128);   // (12, 64)
    proj_gemm_kernel<<<g3, 256, 0, stream>>>(Ao, wproj, out);
}
